// SAGE_28853590295254
// MI455X (gfx1250) — compile-verified, matrix operand order checked by reading, not run on hardware
//
#include <hip/hip_runtime.h>
#include <hip/hip_bf16.h>

// ---------------------------------------------------------------------------
// GraphSAGE (3 layers, mean aggregation) for MI455X / gfx1250.
//
// Roofline: edge aggregation dominates. v2 data path:
//   - activations live in f16 only between layers (GEMM stores f16 in place)
//   - edge gather reads f16 rows (256B/edge vs 512B), widened to f32 for the
//     atomic segment-sum (precision kept in f32; accumulator is L2-resident)
//   - GEMMs run on v_wmma_f32_16x16x32_f16 with f32 accumulate; both the
//     self and neighbor products chain into the same accumulator tiles
// ---------------------------------------------------------------------------

typedef __attribute__((ext_vector_type(16))) _Float16 v16h;
typedef __attribute__((ext_vector_type(4)))  _Float16 v4h;
typedef __attribute__((ext_vector_type(8)))  float    v8f;

#define D_IN 128   // feature width of every layer's input

// ---------------- utility kernels ----------------

__global__ void zero_f32(float* __restrict__ p, int n) {
    int i = blockIdx.x * blockDim.x + threadIdx.x;
    if (i < n) p[i] = 0.0f;
}

__global__ void degree_accum(const int* __restrict__ dst, float* __restrict__ deg, int E) {
    int e = blockIdx.x * blockDim.x + threadIdx.x;
    if (e < E) atomicAdd(deg + dst[e], 1.0f);
}

__global__ void finalize_invdeg(float* __restrict__ deg, int N) {
    int i = blockIdx.x * blockDim.x + threadIdx.x;
    if (i < N) deg[i] = 1.0f / fmaxf(deg[i], 1.0f);
}

// One wave per edge: 32 lanes x 4 halves = 128 features. Coalesced 8B-per-lane
// gather of the f16 source row, widened to f32, coalesced atomic adds into the
// f32 destination row (accumulator stays resident in the 192MB L2).
__global__ __launch_bounds__(256) void scatter_add_f16(const _Float16* __restrict__ h,
                                                       const int* __restrict__ src,
                                                       const int* __restrict__ dst,
                                                       float* __restrict__ agg, int E) {
    const int e = blockIdx.x * 8 + (threadIdx.x >> 5);
    if (e >= E) return;
    const int c = (threadIdx.x & 31) * 4;
    const v4h v = *(const v4h*)(h + (size_t)src[e] * D_IN + c);
    float* p = agg + (size_t)dst[e] * D_IN + c;
    atomicAdd(p + 0, (float)v[0]);
    atomicAdd(p + 1, (float)v[1]);
    atomicAdd(p + 2, (float)v[2]);
    atomicAdd(p + 3, (float)v[3]);
}

__global__ void f32_to_f16(const float* __restrict__ in, _Float16* __restrict__ out, int n) {
    int i = blockIdx.x * blockDim.x + threadIdx.x;
    if (i < n) out[i] = (_Float16)in[i];
}

// agg16[n][c] = f16(agg32[n][c] * inv_deg[n]);  D_IN == 128 -> shifts
__global__ void norm_to_f16(const float* __restrict__ agg, const float* __restrict__ invdeg,
                            _Float16* __restrict__ out, int n) {
    int i = blockIdx.x * blockDim.x + threadIdx.x;
    if (i < n) out[i] = (_Float16)(agg[i] * invdeg[i >> 7]);
}

// wt[o][k] = f16(w[k][o])  (w is [K][OUT] row-major, wt is [OUT][K])
__global__ void transpose_w_f16(const float* __restrict__ w, _Float16* __restrict__ wt,
                                int K, int OUT) {
    int i = blockIdx.x * blockDim.x + threadIdx.x;
    if (i < K * OUT) {
        int k = i / OUT, o = i - k * OUT;
        wt[(size_t)o * K + k] = (_Float16)w[i];
    }
}

// ---------------- fused WMMA GEMM ----------------
// out[r,:] = (relu)( h16[r,:] @ Wself + agg16[r,:] @ Wneigh + b )
// One wave computes a 16-row strip across all OUT columns.
// A-frag: lane holds row (lane&15), K-half (lane>>4): contiguous 16 halves.
// B-frag: lane holds col (lane&15), K-half (lane>>4): contiguous 16 halves of
//         the pre-transposed weight row wt[col][.].
// C/D layout (ISA 7.12.2): acc[t] VGPR v, lane l -> M = v + 8*(l>>4),
//                          Ncol = t*16 + (l&15).
// In-place f16 output is safe: each wave reads only its own 16 rows (fully
// consumed in the K loop) before storing those same rows.
template <int OUT, bool RELU, typename OutT>
__global__ __launch_bounds__(128) void sage_gemm_wmma(const _Float16* __restrict__ h16,
                                                      const _Float16* __restrict__ a16,
                                                      const _Float16* __restrict__ wtS,
                                                      const _Float16* __restrict__ wtN,
                                                      const float* __restrict__ bias,
                                                      OutT* __restrict__ out, int N) {
    constexpr int K  = D_IN;
    constexpr int NT = OUT / 16;

    const int wave = threadIdx.x >> 5;
    const int lane = threadIdx.x & 31;
    const int row0 = (blockIdx.x * 4 + wave) * 16;
    if (row0 >= N) return;                 // wave-uniform: EXEC stays all-ones

    const int rlane = lane & 15;
    const int khalf = lane >> 4;
    const size_t r  = (size_t)(row0 + rlane);   // N % 16 == 0 for this problem

    v8f acc[NT];
#pragma unroll
    for (int t = 0; t < NT; ++t) acc[t] = (v8f){0, 0, 0, 0, 0, 0, 0, 0};

#pragma unroll
    for (int kb = 0; kb < K / 32; ++kb) {
        const int kbase = kb * 32 + khalf * 16;
        const v16h aS = *(const v16h*)(h16 + r * K + kbase);
        const v16h aN = *(const v16h*)(a16 + r * K + kbase);
#pragma unroll
        for (int t = 0; t < NT; ++t) {
            const size_t col = (size_t)(t * 16 + rlane);
            const v16h bS = *(const v16h*)(wtS + col * K + kbase);
            const v16h bN = *(const v16h*)(wtN + col * K + kbase);
            acc[t] = __builtin_amdgcn_wmma_f32_16x16x32_f16(
                false, aS, false, bS, (short)0, acc[t], false, false);
            acc[t] = __builtin_amdgcn_wmma_f32_16x16x32_f16(
                false, aN, false, bN, (short)0, acc[t], false, false);
        }
    }

#pragma unroll
    for (int t = 0; t < NT; ++t) {
        const int col  = t * 16 + rlane;
        const float bv = bias[col];
#pragma unroll
        for (int v = 0; v < 8; ++v) {
            float x = acc[t][v] + bv;
            if (RELU) x = fmaxf(x, 0.0f);
            out[(size_t)(row0 + v + 8 * khalf) * OUT + col] = (OutT)x;
        }
    }
}

// ---------------- host launch ----------------

extern "C" void kernel_launch(void* const* d_in, const int* in_sizes, int n_in,
                              void* d_out, int out_size, void* d_ws, size_t ws_size,
                              hipStream_t stream) {
    const float* features  = (const float*)d_in[0];
    const float* Wself[3]  = {(const float*)d_in[1], (const float*)d_in[4], (const float*)d_in[7]};
    const float* Wneigh[3] = {(const float*)d_in[2], (const float*)d_in[5], (const float*)d_in[8]};
    const float* bias[3]   = {(const float*)d_in[3], (const float*)d_in[6], (const float*)d_in[9]};
    const int* edge_src = (const int*)d_in[10];
    const int* edge_dst = (const int*)d_in[11];

    const int N = in_sizes[0] / D_IN;          // 100000 (multiple of 16)
    const int E = in_sizes[10];                // 1600000
    float* out = (float*)d_out;                // [N][64] f32

    // workspace carve-up (256B aligned)
    char*  ws  = (char*)d_ws;
    size_t off = 0;
    auto alloc = [&](size_t bytes) -> void* {
        void* p = ws + off;
        off = (off + bytes + 255) & ~(size_t)255;
        return p;
    };
    _Float16* h16   = (_Float16*)alloc((size_t)N * D_IN * sizeof(_Float16));
    _Float16* agg16 = (_Float16*)alloc((size_t)N * D_IN * sizeof(_Float16));
    float*    agg32 = (float*)   alloc((size_t)N * D_IN * sizeof(float));
    float*    ideg  = (float*)   alloc((size_t)N * sizeof(float));
    _Float16* wtS   = (_Float16*)alloc((size_t)D_IN * D_IN * sizeof(_Float16));
    _Float16* wtN   = (_Float16*)alloc((size_t)D_IN * D_IN * sizeof(_Float16));
    (void)ws_size;

    const int TB  = 256;
    const int ND  = N * D_IN;
    const int gND = (ND + TB - 1) / TB;
    const int gN  = (N + TB - 1) / TB;
    const int gE  = (E + TB - 1) / TB;
    const int gSc = (E + 7) / 8;               // scatter: 1 wave/edge, 8 edges/block
    const int gGm = (N / 16 + 3) / 4;          // gemm: 4 waves/block, 16 rows/wave

    // inverse degree (shared by all layers)
    zero_f32<<<gN, TB, 0, stream>>>(ideg, N);
    degree_accum<<<gE, TB, 0, stream>>>(edge_dst, ideg, E);
    finalize_invdeg<<<gN, TB, 0, stream>>>(ideg, N);

    // layer-0 input: features converted once to f16; thereafter the GEMM
    // keeps h16 up to date in place.
    f32_to_f16<<<gND, TB, 0, stream>>>(features, h16, ND);

    for (int l = 0; l < 3; ++l) {
        const int OUT = (l == 2) ? 64 : 128;
        const int gW  = (D_IN * OUT + TB - 1) / TB;

        transpose_w_f16<<<gW, TB, 0, stream>>>(Wself[l],  wtS, D_IN, OUT);
        transpose_w_f16<<<gW, TB, 0, stream>>>(Wneigh[l], wtN, D_IN, OUT);

        zero_f32<<<gND, TB, 0, stream>>>(agg32, ND);
        scatter_add_f16<<<gSc, TB, 0, stream>>>(h16, edge_src, edge_dst, agg32, E);
        norm_to_f16<<<gND, TB, 0, stream>>>(agg32, ideg, agg16, ND);

        if (l < 2) {
            sage_gemm_wmma<128, true, _Float16><<<gGm, 128, 0, stream>>>(
                h16, agg16, wtS, wtN, bias[l], h16, N);   // in-place f16 update
        } else {
            sage_gemm_wmma<64, false, float><<<gGm, 128, 0, stream>>>(
                h16, agg16, wtS, wtN, bias[l], out, N);
        }
    }
}
